// EncoderDecoderPreisachNNModel_34780645163512
// MI455X (gfx1250) — matrix operand order repaired and checked by reading, not compiled, not run on hardware
//
#include <hip/hip_runtime.h>
#include <hip/hip_bf16.h>
#include <math.h>

// ---------------- problem constants ----------------
#define HID    256
#define NB     16
#define NS     256
#define NT     1024
#define MPTS   5151          // triangle mesh points for MESH_SIZE=0.01
#define MTILES 322           // ceil(5151/16)
#define NLAYER 3

typedef __bf16 bf16_t;
typedef __attribute__((ext_vector_type(16))) __bf16 v16bf;
typedef __attribute__((ext_vector_type(8)))  __bf16 v8bf;
typedef __attribute__((ext_vector_type(8)))  float   v8f;

// ---------------- K0: prep — transpose Wr to bf16 [layer][n][k], broadcast mesh to output
__global__ void prep_kernel(const float* __restrict__ Wr, bf16_t* __restrict__ WrT,
                            const float* __restrict__ base_mesh, float* __restrict__ out_mesh) {
    int idx = blockIdx.x * blockDim.x + threadIdx.x;
    if (idx < NLAYER * HID * HID) {
        int layer = idx / (HID * HID);
        int rem   = idx % (HID * HID);
        int n = rem / HID, k = rem % HID;
        WrT[idx] = (bf16_t)Wr[layer * HID * HID + k * HID + n];
    }
    if (idx < NB * MPTS * 2) {
        out_mesh[idx] = base_mesh[idx % (MPTS * 2)];
    }
}

// ---------------- K1: encoder context  ctx[b][h] = masked mean of relu(enc@Ws+bs)
__global__ void ctx_kernel(const float* __restrict__ enc_in, const float* __restrict__ mask,
                           const float* __restrict__ Ws, const float* __restrict__ bs,
                           float* __restrict__ ctx) {
    int idx = blockIdx.x * blockDim.x + threadIdx.x;
    if (idx >= NB * HID) return;
    int b = idx / HID, h = idx % HID;
    float w0 = Ws[h], w1 = Ws[HID + h], bb = bs[h];
    float acc = 0.f, msum = 0.f;
    for (int s = 0; s < NS; ++s) {
        float e0 = enc_in[(b * NS + s) * 2];
        float e1 = enc_in[(b * NS + s) * 2 + 1];
        float v  = e0 * w0 + e1 * w1 + bb;
        v = v > 0.f ? v : 0.f;
        float mk = mask[b * NS + s];
        acc += v * mk;
        msum += mk;
    }
    ctx[idx] = acc / fmaxf(msum, 1.f);
}

// ---------------- K2: ctxWc[b][n] = ctx[b] @ Wc
__global__ void ctxwc_kernel(const float* __restrict__ ctx, const float* __restrict__ Wc,
                             float* __restrict__ ctxWc) {
    int idx = blockIdx.x * blockDim.x + threadIdx.x;
    if (idx >= NB * HID) return;
    int b = idx / HID, n = idx % HID;
    float acc = 0.f;
    for (int k = 0; k < HID; ++k) acc += ctx[b * HID + k] * Wc[k * HID + n];
    ctxWc[idx] = acc;
}

// ---------------- K3: density MLP with WMMA (bf16 -> f32 accumulate)
// One workgroup (256 thr = 8 waves) per 16-row tile; each wave owns 2 column tiles.
__global__ __launch_bounds__(256) void dens_mlp_kernel(
    const float*  __restrict__ base_mesh,
    const float*  __restrict__ Win, const float* __restrict__ bin,
    const bf16_t* __restrict__ WrT, const float* __restrict__ br,
    const float*  __restrict__ Wout, const float* __restrict__ bout,
    float* __restrict__ density_ws, float* __restrict__ out_density) {

    __shared__ __align__(16) float  xs[16][HID];   // f32 activations (residual source)
    __shared__ __align__(16) bf16_t xa[16][HID];   // bf16 A operand staging

    const int tile = blockIdx.x;
    const int tid  = threadIdx.x;
    const int m0   = tile * 16;

    // stage x0 = relu(mesh @ Win + bin)
    for (int idx = tid; idx < 16 * HID; idx += 256) {
        int r = idx >> 8, c = idx & (HID - 1);
        int m = m0 + r;
        float be = 0.f, al = 0.f;
        if (m < MPTS) { be = base_mesh[2 * m]; al = base_mesh[2 * m + 1]; }
        float v = be * Win[c] + al * Win[HID + c] + bin[c];
        v = v > 0.f ? v : 0.f;
        xs[r][c] = v;
        xa[r][c] = (bf16_t)v;
    }
    __syncthreads();

    const int wave = tid >> 5;       // 0..7
    const int lane = tid & 31;
    const int row  = lane & 15;      // A row / B,C column-in-tile
    const int hlf  = lane >> 4;      // lane half
    const int n0   = wave * 32;      // this wave's two N tiles: n0, n0+16

    for (int layer = 0; layer < NLAYER; ++layer) {
        const bf16_t* Wt   = WrT + layer * HID * HID;     // [n][k]
        const float*  bias = br  + layer * HID;
        v8f acc0 = {0.f,0.f,0.f,0.f,0.f,0.f,0.f,0.f};
        v8f acc1 = {0.f,0.f,0.f,0.f,0.f,0.f,0.f,0.f};

        for (int kk = 0; kk < HID; kk += 32) {
            // A (16x32 bf16): lane half=0 -> K[kk..kk+15]; half=1 -> K[kk+16..kk+31]
            v8bf alo = *(const v8bf*)&xa[row][kk + hlf * 16];
            v8bf ahi = *(const v8bf*)&xa[row][kk + hlf * 16 + 8];
            v16bf a = __builtin_shufflevector(alo, ahi,
                         0,1,2,3,4,5,6,7,8,9,10,11,12,13,14,15);
            // B (32x16 bf16): lane holds N=n, K = kk + hlf*16 + i (contiguous in Wt[n][k])
            v16bf b0 = *(const v16bf*)&Wt[(n0 +      row) * HID + kk + hlf * 16];
            v16bf b1 = *(const v16bf*)&Wt[(n0 + 16 + row) * HID + kk + hlf * 16];
            acc0 = __builtin_amdgcn_wmma_f32_16x16x32_bf16(false, a, false, b0,
                                                           (short)0, acc0, false, false);
            acc1 = __builtin_amdgcn_wmma_f32_16x16x32_bf16(false, a, false, b1,
                                                           (short)0, acc1, false, false);
        }

        // bias + relu + residual (reads xs) before anyone overwrites the buffers
        float nv0[8], nv1[8];
        #pragma unroll
        for (int v = 0; v < 8; ++v) {
            int rr = v + 8 * hlf;                 // output row in tile
            int c0 = n0 + row, c1 = n0 + 16 + row;
            float t0 = acc0[v] + bias[c0]; t0 = t0 > 0.f ? t0 : 0.f;
            float t1 = acc1[v] + bias[c1]; t1 = t1 > 0.f ? t1 : 0.f;
            nv0[v] = xs[rr][c0] + t0;
            nv1[v] = xs[rr][c1] + t1;
        }
        __syncthreads();
        #pragma unroll
        for (int v = 0; v < 8; ++v) {
            int rr = v + 8 * hlf;
            int c0 = n0 + row, c1 = n0 + 16 + row;
            xs[rr][c0] = nv0[v]; xa[rr][c0] = (bf16_t)nv0[v];
            xs[rr][c1] = nv1[v]; xa[rr][c1] = (bf16_t)nv1[v];
        }
        __syncthreads();
    }

    // output head: density = sigmoid(x @ Wout + bout)
    if (tid < 16) {
        int m = m0 + tid;
        if (m < MPTS) {
            float acc = bout[0];
            for (int c = 0; c < HID; ++c) acc += xs[tid][c] * Wout[c];
            float d = 1.f / (1.f + __expf(-acc));
            density_ws[m] = d;
            for (int b = 0; b < NB; ++b) out_density[b * MPTS + m] = d;
        }
    }
}

// ---------------- K4: initial states
__global__ void init_states_kernel(const float* __restrict__ base_mesh,
                                   const float* __restrict__ density_ws,
                                   const float* __restrict__ Wm, const float* __restrict__ bm,
                                   const float* __restrict__ ctxWc,
                                   const float* __restrict__ Wo, const float* __restrict__ bo,
                                   float* __restrict__ out_init) {
    int idx = blockIdx.x * blockDim.x + threadIdx.x;
    if (idx >= NB * MPTS) return;
    int b = idx / MPTS, m = idx % MPTS;
    float be = base_mesh[2 * m], al = base_mesh[2 * m + 1], d = density_ws[m];
    const float* cw = ctxWc + b * HID;
    float acc = bo[0];
    for (int c = 0; c < HID; ++c) {
        float z = be * Wm[c] + al * Wm[HID + c] + d * Wm[2 * HID + c] + cw[c] + bm[c];
        z = z > 0.f ? z : 0.f;
        acc += z * Wo[c];
    }
    out_init[idx] = tanhf(acc);
}

// ---------------- K5: sequential relay scan, one workgroup per batch
__device__ inline float block_reduce_1024(float v, float* red, int tid) {
    #pragma unroll
    for (int off = 16; off; off >>= 1) v += __shfl_xor(v, off, 32);
    int wid = tid >> 5, lane = tid & 31;
    if (lane == 0) red[wid] = v;
    __syncthreads();
    if (wid == 0) {
        float x = red[lane];
        #pragma unroll
        for (int off = 16; off; off >>= 1) x += __shfl_xor(x, off, 32);
        if (lane == 0) red[0] = x;
    }
    __syncthreads();
    float r = red[0];
    __syncthreads();
    return r;
}

__global__ __launch_bounds__(1024) void relay_kernel(
    const float* __restrict__ base_mesh, const float* __restrict__ dec_in,
    const float* __restrict__ density_ws, const float* __restrict__ init_states,
    const float* __restrict__ hsr, const float* __restrict__ msr, const float* __restrict__ mor,
    float* __restrict__ out_b, float* __restrict__ out_m) {

    const int b   = blockIdx.x;
    const int tid = threadIdx.x;
    __shared__ float hbuf[NT];
    __shared__ float red[32];

    hbuf[tid] = dec_in[b * NT + tid];

    float s[6], dn[6], al[6], be[6];
    float dpart = 0.f;
    #pragma unroll
    for (int k = 0; k < 6; ++k) {
        int m = tid + k * 1024;
        if (m < MPTS) {
            s[k]  = init_states[b * MPTS + m];
            dn[k] = density_ws[m];
            be[k] = base_mesh[2 * m];
            al[k] = base_mesh[2 * m + 1];
            dpart += dn[k];
        } else { s[k] = 0.f; dn[k] = 0.f; al[k] = 0.f; be[k] = 0.f; }
    }

    float dsum = block_reduce_1024(dpart, red, tid);   // also orders hbuf writes

    // interval-mapped scalars: sigmoid to [0,10] / [-10,10]
    float hs = 10.f / (1.f + __expf(-hsr[0]));
    float ms = 10.f / (1.f + __expf(-msr[0]));
    float mo = -10.f + 20.f / (1.f + __expf(-mor[0]));

    for (int t = 0; t < NT; ++t) {
        float ht = hbuf[t];
        float part = 0.f;
        #pragma unroll
        for (int k = 0; k < 6; ++k) {
            float wup = 1.f / (1.f + __expf(-(ht - al[k]) * 1000.f));  // /TEMP
            float wdn = 1.f / (1.f + __expf(-(be[k] - ht) * 1000.f));
            float sv = s[k];
            sv = sv + wup * (1.f - sv);
            sv = sv + wdn * (-1.f - sv);
            s[k] = sv;
            part += dn[k] * sv;
        }
        float tot = block_reduce_1024(part, red, tid);
        if (tid == 0) {
            float mt = tot / dsum;
            out_m[b * NT + t] = mt;
            out_b[b * NT + t] = hs * ht + ms * mt + mo;
        }
    }
}

// ---------------- launch ----------------
extern "C" void kernel_launch(void* const* d_in, const int* in_sizes, int n_in,
                              void* d_out, int out_size, void* d_ws, size_t ws_size,
                              hipStream_t stream) {
    const float* enc_in = (const float*)d_in[0];
    const float* dec_in = (const float*)d_in[1];
    const float* mask   = (const float*)d_in[2];
    const float* mesh   = (const float*)d_in[3];
    const float* Win    = (const float*)d_in[4];
    const float* bin    = (const float*)d_in[5];
    const float* Wr     = (const float*)d_in[6];
    const float* br     = (const float*)d_in[7];
    const float* Wout   = (const float*)d_in[8];
    const float* bout   = (const float*)d_in[9];
    const float* Ws     = (const float*)d_in[10];
    const float* bs     = (const float*)d_in[11];
    const float* Wm     = (const float*)d_in[12];
    const float* Wc     = (const float*)d_in[13];
    const float* bm     = (const float*)d_in[14];
    const float* Wo     = (const float*)d_in[15];
    const float* bo     = (const float*)d_in[16];
    const float* hsr    = (const float*)d_in[17];
    const float* msr    = (const float*)d_in[18];
    const float* mor    = (const float*)d_in[19];

    float* out      = (float*)d_out;
    float* out_b    = out;                         // (16,1024)
    float* out_dens = out + NB * NT;               // (16,5151)
    float* out_m    = out + NB * NT + NB * MPTS;   // (16,1024)
    float* out_init = out_m + NB * NT;             // (16,5151)
    float* out_mesh = out_init + NB * MPTS;        // (16,5151,2)

    char*   ws      = (char*)d_ws;
    bf16_t* WrT     = (bf16_t*)ws;                         // 3*256*256 bf16 = 393216 B
    float*  dens_ws = (float*)(ws + 393216);               // 5152 f32
    float*  ctx     = (float*)(ws + 413824);               // 16*256 f32
    float*  ctxWc   = (float*)(ws + 430208);               // 16*256 f32

    prep_kernel<<<768, 256, 0, stream>>>(Wr, WrT, mesh, out_mesh);
    ctx_kernel<<<16, 256, 0, stream>>>(enc_in, mask, Ws, bs, ctx);
    ctxwc_kernel<<<16, 256, 0, stream>>>(ctx, Wc, ctxWc);
    dens_mlp_kernel<<<MTILES, 256, 0, stream>>>(mesh, Win, bin, WrT, br, Wout, bout,
                                                dens_ws, out_dens);
    init_states_kernel<<<(NB * MPTS + 255) / 256, 256, 0, stream>>>(
        mesh, dens_ws, Wm, bm, ctxWc, Wo, bo, out_init);
    relay_kernel<<<NB, 1024, 0, stream>>>(mesh, dec_in, dens_ws, out_init,
                                          hsr, msr, mor, out_b, out_m);
}
